// MambaLayer_18665927868805
// MI455X (gfx1250) — matrix operand order repaired and checked by reading, not compiled, not run on hardware
//
#include <hip/hip_runtime.h>
#include <hip/hip_bf16.h>

// ---------------------------------------------------------------------------
// Mamba block for MI455X (gfx1250): bf16 WMMA GEMMs + chunked selective scan.
// ---------------------------------------------------------------------------

typedef __attribute__((ext_vector_type(16))) __bf16 v16bf;
typedef __attribute__((ext_vector_type(8)))  float  v8f;

static constexpr int  kB    = 2;
static constexpr int  kDim  = 256;     // d_model
static constexpr int  kDi   = 512;     // d_inner
static constexpr int  kL    = 32768;   // sequence length (32*32*32)
static constexpr int  kNBL  = kB * kL; // 65536 rows
static constexpr int  kS    = 16;      // d_state
static constexpr int  kR    = 16;      // dt_rank
static constexpr int  kNCH  = 128;     // scan chunks per sequence
static constexpr int  kCL   = kL / kNCH; // 256 steps per chunk

#define DEVINL __device__ __forceinline__

DEVINL __bf16 f2bf(float x) { return (__bf16)x; }
DEVINL float  bf2f(__bf16 x) { return (float)x; }
DEVINL float  siluf(float x) { return x / (1.0f + __expf(-x)); }

// spatial permute: l = w*1024 + h*32 + d  ->  offset d*1024 + h*32 + w
DEVINL int perm_spatial(int l) {
    return ((l & 31) << 10) | (((l >> 5) & 31) << 5) | (l >> 10);
}

// ---------------------------------------------------------------------------
// K0: convert f32 weights to bf16 once (tiny).
// ---------------------------------------------------------------------------
__global__ void cvt_weights_kernel(const float* wIn, const float* wX, const float* wOut,
                                   __bf16* bIn, __bf16* bX, __bf16* bOut) {
    int i = blockIdx.x * 256 + threadIdx.x;
    if (i < 2 * kDi * kDim) bIn[i] = f2bf(wIn[i]);     // 1024x256
    if (i < 48 * kDi)       bX[i]  = f2bf(wX[i]);      // 48x512
    if (i < kDim * kDi)     bOut[i] = f2bf(wOut[i]);   // 256x512
}

// ---------------------------------------------------------------------------
// K1: gather (B,C,D,H,W)->(BL,C) + LayerNorm over C, emit bf16 activations.
// ---------------------------------------------------------------------------
__global__ void gather_ln_kernel(const float* __restrict__ x,
                                 const float* __restrict__ lnw,
                                 const float* __restrict__ lnb,
                                 __bf16* __restrict__ xn) {
    int bl = blockIdx.x;            // 0..65535
    int c  = threadIdx.x;           // 0..255
    int b  = bl >> 15, l = bl & (kL - 1);
    int sp = perm_spatial(l);
    float v = x[((size_t)(b * kDim + c) << 15) + sp];

    __shared__ float s1[kDim], s2[kDim];
    s1[c] = v; s2[c] = v * v;
    __syncthreads();
    #pragma unroll
    for (int off = 128; off > 0; off >>= 1) {
        if (c < off) { s1[c] += s1[c + off]; s2[c] += s2[c + off]; }
        __syncthreads();
    }
    float mean = s1[0] * (1.0f / kDim);
    float var  = s2[0] * (1.0f / kDim) - mean * mean;
    float nrm  = (v - mean) * rsqrtf(var + 1e-5f) * lnw[c] + lnb[c];
    xn[(size_t)bl * kDim + c] = f2bf(nrm);
}

// ---------------------------------------------------------------------------
// WMMA fragment loaders (ISA 7.12.2 layouts, wave32).
// A (MxK, 16-bit): lane=row (lane&15); elems 0..7 = K[k0+half*8 ..],
//                  elems 8..15 = K[k0+16+half*8 ..]   (half = lane>>4)
// B (KxN from row-major W[N][K]): lane=col; same K layout as A
// ---------------------------------------------------------------------------
DEVINL v16bf load_frag_a(const __bf16* row, int k0, int half) {
    v16bf f;
    const float4* p0 = reinterpret_cast<const float4*>(row + k0 + (half << 3));
    const float4* p1 = reinterpret_cast<const float4*>(row + k0 + 16 + (half << 3));
    reinterpret_cast<float4*>(&f)[0] = *p0;
    reinterpret_cast<float4*>(&f)[1] = *p1;
    return f;
}
DEVINL v16bf load_frag_b(const __bf16* row, int k0, int half) {
    v16bf f;
    const float4* p0 = reinterpret_cast<const float4*>(row + k0 + (half << 3));
    const float4* p1 = reinterpret_cast<const float4*>(row + k0 + 16 + (half << 3));
    reinterpret_cast<float4*>(&f)[0] = *p0;
    reinterpret_cast<float4*>(&f)[1] = *p1;
    return f;
}

enum { MODE_XZ = 0, MODE_DBC = 1, MODE_OUT = 2 };

// ---------------------------------------------------------------------------
// K2: C = A(bf16, MxK) @ W(bf16, NxK)^T via v_wmma_f32_16x16x32_bf16.
// Block = 4 waves; each wave owns a 16 x (16*NT) tile: the A fragment is
// loaded once per K-step and reused for NT WMMAs (2 + 2*NT b128 loads per
// NT WMMAs instead of 4 per WMMA). Fused epilogues:
//   MODE_XZ : n<512 -> xs f32 ; n>=512 -> silu(z) bf16
//   MODE_DBC: n<16 -> dt ; n<32 -> B ; else C   (all f32)
//   MODE_OUT: scatter f32 into (B,C,D,H,W) output
// ---------------------------------------------------------------------------
template <int MODE, int NT>
__global__ void wmma_gemm_kernel(const __bf16* __restrict__ A,
                                 const __bf16* __restrict__ W, int K,
                                 float* __restrict__ f0, float* __restrict__ f1,
                                 float* __restrict__ f2, __bf16* __restrict__ b0) {
    const int wave = threadIdx.x >> 5;
    const int lane = threadIdx.x & 31;
    const int half = lane >> 4;
    const int rc   = lane & 15;
    const int mtile  = (blockIdx.x * 4 + wave) * 16;
    const int ntile0 = blockIdx.y * (16 * NT);

    const __bf16* arow = A + (size_t)(mtile + rc) * K;
    const __bf16* wrow[NT];
    #pragma unroll
    for (int t = 0; t < NT; ++t)
        wrow[t] = W + (size_t)(ntile0 + t * 16 + rc) * K;

    __builtin_prefetch(arow, 0, 3);   // global_prefetch_b8: warm WGP$ for the K-loop

    v8f acc[NT] = {};
    for (int k0 = 0; k0 < K; k0 += 32) {
        v16bf a = load_frag_a(arow, k0, half);
        #pragma unroll
        for (int t = 0; t < NT; ++t) {
            v16bf b = load_frag_b(wrow[t], k0, half);
            acc[t] = __builtin_amdgcn_wmma_f32_16x16x32_bf16(
                /*neg_a=*/false, a, /*neg_b=*/false, b,
                /*c_mod=*/(short)0, acc[t], /*reuse_a=*/false, /*reuse_b=*/false);
        }
    }

    #pragma unroll
    for (int t = 0; t < NT; ++t) {
        const int ng = ntile0 + t * 16 + rc;   // output column
        #pragma unroll
        for (int e = 0; e < 8; ++e) {          // rows e + 8*half
            const int mg = mtile + e + (half << 3);
            const float v = acc[t][e];
            if (MODE == MODE_XZ) {
                if (ng < kDi) f0[(size_t)mg * kDi + ng] = v;                        // xs
                else          b0[(size_t)mg * kDi + (ng - kDi)] = f2bf(siluf(v));   // silu(z)
            } else if (MODE == MODE_DBC) {
                if (ng < kR)            f0[(size_t)mg * kR + ng] = v;               // dt
                else if (ng < kR + kS)  f1[(size_t)mg * kS + (ng - kR)] = v;        // B_t
                else                    f2[(size_t)mg * kS + (ng - kR - kS)] = v;   // C_t
            } else { // MODE_OUT: scatter to (B,C,D,H,W)
                const int b = mg >> 15, l = mg & (kL - 1);
                f0[((size_t)(b * kDim + ng) << 15) + perm_spatial(l)] = v;
            }
        }
    }
}

// ---------------------------------------------------------------------------
// K3: causal depthwise conv1d (k=4) + bias + SiLU -> u (f32 for scan, bf16 for GEMM)
// ---------------------------------------------------------------------------
__global__ void conv_silu_kernel(const float* __restrict__ xs,
                                 const float* __restrict__ cw,
                                 const float* __restrict__ cb,
                                 float* __restrict__ u, __bf16* __restrict__ ubf) {
    size_t idx = (size_t)blockIdx.x * 256 + threadIdx.x;   // over NBL*Di
    int d = (int)(idx & (kDi - 1));
    size_t bl = idx >> 9;
    int l = (int)(bl & (kL - 1));
    float acc = cb[d];
    #pragma unroll
    for (int j = 0; j < 4; ++j) {
        int lj = l - 3 + j;
        if (lj >= 0) acc += xs[(bl - 3 + j) * kDi + d] * cw[d * 4 + j];
    }
    float s = siluf(acc);
    u[idx] = s;
    ubf[idx] = f2bf(s);
}

// ---------------------------------------------------------------------------
// K4: delta = softplus(dt @ dt_proj_w^T + bias)   (K=16, VALU)
// ---------------------------------------------------------------------------
__global__ void delta_kernel(const float* __restrict__ dtr,
                             const float* __restrict__ dtw,
                             const float* __restrict__ dtb,
                             float* __restrict__ delta) {
    int bl = blockIdx.x;
    int d  = threadIdx.x;             // 512 threads
    __shared__ float r[kR];
    if (d < kR) r[d] = dtr[(size_t)bl * kR + d];
    __syncthreads();
    float acc = dtb[d];
    #pragma unroll
    for (int j = 0; j < kR; ++j) acc += r[j] * dtw[d * kR + j];
    float sp = (acc > 20.0f) ? acc : log1pf(__expf(acc));
    delta[(size_t)bl * kDi + d] = sp;
}

// ---------------------------------------------------------------------------
// Chunked selective scan. The recurrence h_t = exp(dt*A)h_{t-1} + dt*u*B_t
// composes exactly over a chunk: h_end = exp(A * sum(dt)) * h_start + h_local.
// lane = (b, chunk, d), 16 states in registers -> 131072 lanes / 4096 waves.
// ---------------------------------------------------------------------------
__global__ void scan_pass1_kernel(const float* __restrict__ delta,
                                  const float* __restrict__ u,
                                  const float* __restrict__ Bm,
                                  const float* __restrict__ Alog,
                                  float* __restrict__ Ssum,
                                  float* __restrict__ Hend) {
    int tid = blockIdx.x * 256 + threadIdx.x;   // b*65536 + ch*512 + d
    int d = tid & (kDi - 1), ch = (tid >> 9) & (kNCH - 1), b = tid >> 16;
    float A[kS], h[kS];
    #pragma unroll
    for (int s = 0; s < kS; ++s) { A[s] = -__expf(Alog[d * kS + s]); h[s] = 0.0f; }
    float dsum = 0.0f;
    size_t bl0 = ((size_t)b << 15) + (size_t)ch * kCL;
    for (int t = 0; t < kCL; ++t) {
        size_t bl = bl0 + t;
        float dt = delta[bl * kDi + d];
        float du = dt * u[bl * kDi + d];
        dsum += dt;
        const float* Bp = Bm + bl * kS;
        #pragma unroll
        for (int s = 0; s < kS; ++s) h[s] = __expf(dt * A[s]) * h[s] + du * Bp[s];
    }
    Ssum[tid] = dsum;
    #pragma unroll
    for (int s = 0; s < kS; ++s) Hend[(size_t)tid * kS + s] = h[s];
}

__global__ void scan_combine_kernel(const float* __restrict__ Alog,
                                    const float* __restrict__ Ssum,
                                    const float* __restrict__ Hend,
                                    float* __restrict__ Hstart) {
    int tid = blockIdx.x * 256 + threadIdx.x;   // over B*Di*S = 16384
    int s = tid & (kS - 1), d = (tid >> 4) & (kDi - 1), b = tid >> 13;
    float A = -__expf(Alog[d * kS + s]);
    float h = 0.0f;
    for (int ch = 0; ch < kNCH; ++ch) {
        size_t base = (size_t)(b * kNCH + ch) * kDi + d;
        Hstart[base * kS + s] = h;
        h = __expf(A * Ssum[base]) * h + Hend[base * kS + s];
    }
}

__global__ void scan_pass2_kernel(const float* __restrict__ delta,
                                  const float* __restrict__ u,
                                  const float* __restrict__ Bm,
                                  const float* __restrict__ Cm,
                                  const float* __restrict__ Alog,
                                  const float* __restrict__ Dp,
                                  const __bf16* __restrict__ zsil,
                                  const float* __restrict__ Hstart,
                                  __bf16* __restrict__ g) {
    int tid = blockIdx.x * 256 + threadIdx.x;
    int d = tid & (kDi - 1), ch = (tid >> 9) & (kNCH - 1), b = tid >> 16;
    float A[kS], h[kS];
    #pragma unroll
    for (int s = 0; s < kS; ++s) {
        A[s] = -__expf(Alog[d * kS + s]);
        h[s] = Hstart[(size_t)tid * kS + s];
    }
    const float Dd = Dp[d];
    size_t bl0 = ((size_t)b << 15) + (size_t)ch * kCL;
    for (int t = 0; t < kCL; ++t) {
        size_t bl = bl0 + t;
        float dt = delta[bl * kDi + d];
        float uu = u[bl * kDi + d];
        float du = dt * uu;
        const float* Bp = Bm + bl * kS;
        const float* Cp = Cm + bl * kS;
        float y = 0.0f;
        #pragma unroll
        for (int s = 0; s < kS; ++s) {
            h[s] = __expf(dt * A[s]) * h[s] + du * Bp[s];
            y += h[s] * Cp[s];
        }
        y += Dd * uu;
        g[bl * kDi + d] = f2bf(y * bf2f(zsil[bl * kDi + d]));  // gate already silu'd
    }
}

// ---------------------------------------------------------------------------
extern "C" void kernel_launch(void* const* d_in, const int* in_sizes, int n_in,
                              void* d_out, int out_size, void* d_ws, size_t ws_size,
                              hipStream_t stream) {
    const float* x        = (const float*)d_in[0];
    const float* ln_w     = (const float*)d_in[1];
    const float* ln_b     = (const float*)d_in[2];
    const float* in_w     = (const float*)d_in[3];
    const float* conv_w   = (const float*)d_in[4];
    const float* conv_b   = (const float*)d_in[5];
    const float* xproj_w  = (const float*)d_in[6];
    const float* dt_w     = (const float*)d_in[7];
    const float* dt_b     = (const float*)d_in[8];
    const float* A_log    = (const float*)d_in[9];
    const float* D_param  = (const float*)d_in[10];
    const float* out_w    = (const float*)d_in[11];
    float* out = (float*)d_out;

    char* ws = (char*)d_ws;
    size_t off = 0;
    auto take = [&](size_t bytes) -> void* {
        void* p = ws + off;
        off = (off + bytes + 255) & ~(size_t)255;
        return p;
    };
    __bf16* xn    = (__bf16*)take((size_t)kNBL * kDim * 2);
    __bf16* wInB  = (__bf16*)take((size_t)2 * kDi * kDim * 2);
    __bf16* wXB   = (__bf16*)take((size_t)48 * kDi * 2);
    __bf16* wOB   = (__bf16*)take((size_t)kDim * kDi * 2);
    float*  xs    = (float*) take((size_t)kNBL * kDi * 4);
    __bf16* zsil  = (__bf16*)take((size_t)kNBL * kDi * 2);
    float*  u     = (float*) take((size_t)kNBL * kDi * 4);
    __bf16* ubf   = (__bf16*)take((size_t)kNBL * kDi * 2);
    float*  dtr   = (float*) take((size_t)kNBL * kR * 4);
    float*  Bm    = (float*) take((size_t)kNBL * kS * 4);
    float*  Cm    = (float*) take((size_t)kNBL * kS * 4);
    float*  delta = (float*) take((size_t)kNBL * kDi * 4);
    float*  Ssum  = (float*) take((size_t)kB * kNCH * kDi * 4);
    float*  Hend  = (float*) take((size_t)kB * kNCH * kDi * kS * 4);
    float*  Hst   = (float*) take((size_t)kB * kNCH * kDi * kS * 4);
    __bf16* g     = (__bf16*)take((size_t)kNBL * kDi * 2);

    // K0: weight conversion
    cvt_weights_kernel<<<1024, 256, 0, stream>>>(in_w, xproj_w, out_w, wInB, wXB, wOB);
    // K1: gather + LN
    gather_ln_kernel<<<kNBL, kDim, 0, stream>>>(x, ln_w, ln_b, xn);
    // K2a: in_proj GEMM (65536x256 @ 256x1024) -> split xs / silu(z)
    wmma_gemm_kernel<MODE_XZ, 4><<<dim3(kNBL / 64, (2 * kDi) / 64), 128, 0, stream>>>(
        xn, wInB, kDim, xs, nullptr, nullptr, zsil);
    // K3: causal depthwise conv + SiLU
    conv_silu_kernel<<<(kNBL * kDi) / 256, 256, 0, stream>>>(xs, conv_w, conv_b, u, ubf);
    // K2b: x_proj GEMM (65536x512 @ 512x48) -> dt / B / C (one wave covers N=48)
    wmma_gemm_kernel<MODE_DBC, 3><<<dim3(kNBL / 64, 1), 128, 0, stream>>>(
        ubf, wXB, kDi, dtr, Bm, Cm, nullptr);
    // K4: delta = softplus(dt_proj)
    delta_kernel<<<kNBL, kDi, 0, stream>>>(dtr, dt_w, dt_b, delta);
    // K5-7: chunked selective scan
    scan_pass1_kernel<<<(kB * kNCH * kDi) / 256, 256, 0, stream>>>(delta, u, Bm, A_log, Ssum, Hend);
    scan_combine_kernel<<<(kB * kDi * kS) / 256, 256, 0, stream>>>(A_log, Ssum, Hend, Hst);
    scan_pass2_kernel<<<(kB * kNCH * kDi) / 256, 256, 0, stream>>>(
        delta, u, Bm, Cm, A_log, D_param, zsil, Hst, g);
    // K2c: out_proj GEMM (65536x512 @ 512x256) with permuted scatter to output
    wmma_gemm_kernel<MODE_OUT, 4><<<dim3(kNBL / 64, kDim / 64), 128, 0, stream>>>(
        g, wOB, kDi, out, nullptr, nullptr, nullptr);
}
